// GraphRes_70909910057698
// MI455X (gfx1250) — matrix operand order repaired and checked by reading, not compiled, not run on hardware
//
#include <hip/hip_runtime.h>
#include <hip/hip_bf16.h>
#include <math.h>

typedef __attribute__((ext_vector_type(16))) _Float16 v16h;
typedef __attribute__((ext_vector_type(8)))  _Float16 v8h;
typedef __attribute__((ext_vector_type(8)))  float    v8f;

#define KK 27   // 3^3 spline kernels

__device__ __forceinline__ float eluf(float x) { return x > 0.f ? x : (__expf(x) - 1.f); }

__device__ __forceinline__ void atomicMaxF(float* addr, float v) {
  if (v >= 0.f) atomicMax((int*)addr, __float_as_int(v));
  else          atomicMin((unsigned int*)addr, __float_as_uint(v));
}

// packed f16 atomic add: CDNA5 GLOBAL_ATOMIC_PK_ADD_F16 (no builtin -> inline asm)
__device__ __forceinline__ void atomicPkAddF16(_Float16* addr, float v0, float v1) {
  union { _Float16 h[2]; unsigned int u; } pk;
  pk.h[0] = (_Float16)v0;
  pk.h[1] = (_Float16)v1;
  unsigned long long a = (unsigned long long)addr;
  asm volatile("global_atomic_pk_add_f16 %0, %1, off"
               :: "v"(a), "v"(pk.u) : "memory");
}

__device__ __forceinline__ void spline_basis(const float* __restrict__ ea, int e,
                                             float frac[3], int ilo[3]) {
#pragma unroll
  for (int d = 0; d < 3; ++d) {
    float v  = ea[e * 3 + d] * 2.f;            // (m-1) = 2
    float lo = fminf(fmaxf(floorf(v), 0.f), 1.f);
    frac[d]  = v - lo;
    ilo[d]   = (int)lo;
  }
}

// ---------------- layer 1: scatter (fin = 1) ----------------
__global__ void k_spline1(const float* __restrict__ ea, const float* __restrict__ x,
                          const int* __restrict__ ei, float* __restrict__ acc1,
                          float* __restrict__ deg, int E) {
  int e = blockIdx.x * blockDim.x + threadIdx.x;
  if (e >= E) return;
  int src = ei[e], dst = ei[E + e];
  float frac[3]; int ilo[3];
  spline_basis(ea, e, frac, ilo);
  float xj = x[src];
  atomicAdd(&deg[dst], 1.0f);
#pragma unroll
  for (int b = 0; b < 8; ++b) {
    float w = ((b & 1) ? frac[0] : 1.f - frac[0]) *
              ((b & 2) ? frac[1] : 1.f - frac[1]) *
              ((b & 4) ? frac[2] : 1.f - frac[2]);
    int kidx = (ilo[0] + (b & 1)) + 3 * (ilo[1] + ((b >> 1) & 1)) + 9 * (ilo[2] + ((b >> 2) & 1));
    atomicAdd(&acc1[dst * KK + kidx], w * xj);
  }
}

// ---------------- layer 1: per-node matmul (27x32) + ELU + BN stats ----------------
__global__ void k_node1(const float* __restrict__ acc1, const float* __restrict__ deg,
                        const float* __restrict__ W1, float* __restrict__ h1,
                        float* __restrict__ bsum, float* __restrict__ bsq, int N) {
  __shared__ float W1s[KK * 32];
  __shared__ float red[256];
  int tid = threadIdx.x;
  for (int i = tid; i < KK * 32; i += 256) W1s[i] = W1[i];
  __syncthreads();
  int c   = tid & 31;
  int row = tid >> 5;
  int n   = blockIdx.x * 8 + row;
  float h = 0.f;
  if (n < N) {
    float s = 0.f;
#pragma unroll
    for (int k = 0; k < KK; ++k) s += acc1[n * KK + k] * W1s[k * 32 + c];
    float d = fmaxf(deg[n], 1.f);
    h = eluf(s / d);
    h1[n * 32 + c] = h;
  }
  red[tid] = h;
  __syncthreads();
  if (row == 0) {
    float s = 0.f;
#pragma unroll
    for (int r = 0; r < 8; ++r) s += red[r * 32 + c];
    atomicAdd(&bsum[c], s);
  }
  __syncthreads();
  red[tid] = h * h;
  __syncthreads();
  if (row == 0) {
    float s = 0.f;
#pragma unroll
    for (int r = 0; r < 8; ++r) s += red[r * 32 + c];
    atomicAdd(&bsq[c], s);
  }
}

// ---------------- batchnorm: finalize to scale/shift ----------------
__global__ void k_bn_fin(const float* __restrict__ bsum, const float* __restrict__ bsq,
                         const float* __restrict__ g, const float* __restrict__ b,
                         float* __restrict__ a_out, float* __restrict__ b_out,
                         int C, float invN) {
  int c = threadIdx.x;
  if (c >= C) return;
  float mu  = bsum[c] * invN;
  float var = bsq[c] * invN - mu * mu;
  float is  = rsqrtf(var + 1e-5f);
  float a   = g[c] * is;
  a_out[c]  = a;
  b_out[c]  = b[c] - mu * a;
}

__global__ void k_bn_apply(float* __restrict__ h, const float* __restrict__ a,
                           const float* __restrict__ bb, int total, int Cmask) {
  int i = blockIdx.x * blockDim.x + threadIdx.x;
  if (i >= total) return;
  int c = i & Cmask;
  h[i] = h[i] * a[c] + bb[c];
}

// ---------------- pack W2 (f32 [864][128]) into B-fragment-ordered f16 ----------------
// read index in gemm: W2h[(((kb*8+nt)*32+lane)*16)+j]  == B[k = kb*32+kbase+j][n = nt*16+(lane&15)]
__global__ void k_w2pack(const float* __restrict__ W2, _Float16* __restrict__ W2h) {
  int idx = blockIdx.x * 256 + threadIdx.x;
  if (idx >= KK * 32 * 128) return;        // 110592
  int j    = idx & 15;
  int lane = (idx >> 4) & 31;
  int nt   = (idx >> 9) & 7;
  int kb   = idx >> 12;                    // 0..26
  int kbase = (lane < 16) ? 0 : 16;
  int k = kb * 32 + kbase + j;             // 0..863
  int n = nt * 16 + (lane & 15);           // 0..127
  W2h[idx] = (_Float16)W2[k * 128 + n];
}

// ---------------- layer 2: scatter with packed-f16 atomics ----------------
// one wave = 2 edges; 16 lanes per edge; each lane owns feature pair (2f, 2f+1)
__global__ void k_spline2(const float* __restrict__ ea, const float* __restrict__ h1,
                          const int* __restrict__ ei, _Float16* __restrict__ acc2, int E) {
  int gt    = blockIdx.x * blockDim.x + threadIdx.x;
  int lane  = gt & 31;
  int fpair = lane & 15;                  // feature pair index 0..15
  int e     = ((gt >> 5) << 1) + (lane >> 4);
  if (e >= E) return;
  int src = ei[e], dst = ei[E + e];
  float frac[3]; int ilo[3];
  spline_basis(ea, e, frac, ilo);
  const float2* hp = (const float2*)(h1 + (size_t)src * 32 + fpair * 2);
  float2 hv = *hp;
#pragma unroll
  for (int b = 0; b < 8; ++b) {
    float w = ((b & 1) ? frac[0] : 1.f - frac[0]) *
              ((b & 2) ? frac[1] : 1.f - frac[1]) *
              ((b & 4) ? frac[2] : 1.f - frac[2]);
    int kidx = (ilo[0] + (b & 1)) + 3 * (ilo[1] + ((b >> 1) & 1)) + 9 * (ilo[2] + ((b >> 2) & 1));
    atomicPkAddF16(acc2 + ((size_t)(dst * KK + kidx) * 32 + fpair * 2), w * hv.x, w * hv.y);
  }
}

// ---------------- layer 2: WMMA GEMM  h2[N,128] = acc2[N,864](f16) x W2h[864,128] ----------------
// block = 128 threads = 4 waves; wave owns 16 rows x 128 cols (8 accumulators)
__global__ void __launch_bounds__(128)
k_gemm2(const _Float16* __restrict__ acc2, const _Float16* __restrict__ W2h,
        const float* __restrict__ deg, float* __restrict__ h2,
        float* __restrict__ bsum, float* __restrict__ bsq, int N) {
  int lane    = threadIdx.x & 31;
  int wid     = threadIdx.x >> 5;
  int rowbase = blockIdx.x * 64 + wid * 16;
  if (rowbase >= N) return;                 // wave-uniform: EXEC stays all-ones

  const v8f zero = {0.f, 0.f, 0.f, 0.f, 0.f, 0.f, 0.f, 0.f};
  v8f acc[8];
#pragma unroll
  for (int t = 0; t < 8; ++t) acc[t] = zero;

  int r  = rowbase + (lane & 15);
  int ko = (lane < 16) ? 0 : 16;
  const _Float16* arow = acc2 + (size_t)r * 864;

  for (int kb = 0; kb < 27; ++kb) {
    v16h a;
    if (r < N) {
      // two 16-byte loads: K = kb*32 + {ko..ko+15} (contiguous), no format conversion
      v8h lo = *(const v8h*)(arow + kb * 32 + ko);
      v8h hi = *(const v8h*)(arow + kb * 32 + ko + 8);
      a = __builtin_shufflevector(lo, hi, 0, 1, 2, 3, 4, 5, 6, 7,
                                          8, 9, 10, 11, 12, 13, 14, 15);
    } else {
#pragma unroll
      for (int j = 0; j < 16; ++j) a[j] = (_Float16)0.f;
    }

    // preload all 8 B fragments (clause-able), then issue the 8-WMMA chain
    v16h bfr[8];
#pragma unroll
    for (int nt = 0; nt < 8; ++nt)
      bfr[nt] = *(const v16h*)(W2h + ((size_t)(kb * 8 + nt) * 32 + lane) * 16);
#pragma unroll
    for (int nt = 0; nt < 8; ++nt)
      acc[nt] = __builtin_amdgcn_wmma_f32_16x16x32_f16(
          false, a, false, bfr[nt], (short)0, acc[nt], false, false);
  }

  // epilogue: /deg, ELU, store, BN stats
  int mofs = (lane >> 4) * 8;
  float invd[8];
#pragma unroll
  for (int v = 0; v < 8; ++v) {
    int rr  = rowbase + v + mofs;
    invd[v] = (rr < N) ? 1.f / fmaxf(deg[rr], 1.f) : 0.f;
  }
#pragma unroll
  for (int nt = 0; nt < 8; ++nt) {
    int col = nt * 16 + (lane & 15);
    float s = 0.f, s2 = 0.f;
#pragma unroll
    for (int v = 0; v < 8; ++v) {
      int rr = rowbase + v + mofs;
      if (rr < N) {
        float val = eluf(acc[nt][v] * invd[v]);
        h2[(size_t)rr * 128 + col] = val;
        s += val; s2 += val * val;
      }
    }
    atomicAdd(&bsum[col], s);
    atomicAdd(&bsq[col],  s2);
  }
}

// ---------------- voxel max-pool (BN2 fused) ----------------
__global__ void k_pool(const float* __restrict__ h2, const float* __restrict__ a,
                       const float* __restrict__ bb, const float* __restrict__ pos,
                       const int* __restrict__ batch, float* __restrict__ pooled, int N) {
  int i = blockIdx.x * blockDim.x + threadIdx.x;
  if (i >= N * 128) return;
  int n = i >> 7, c = i & 127;
  float val = h2[i] * a[c] + bb[c];
  int cx = min(max((int)floorf(pos[n * 3 + 0] / 60.f), 0), 3);
  int cy = min(max((int)floorf(pos[n * 3 + 1] / 45.f), 0), 3);
  int cluster = batch[n] * 16 + cy * 4 + cx;
  atomicMaxF(&pooled[cluster * 128 + c], val);
}

__global__ void k_poolfix(float* __restrict__ pooled, int total) {
  int i = blockIdx.x * blockDim.x + threadIdx.x;
  if (i >= total) return;
  float v = pooled[i];
  if (!(v > -3.0e38f && v < 3.0e38f)) pooled[i] = 0.f;   // NaN / +-inf -> 0
}

// ---------------- final FC: out[B,101] = feat[B,2048] @ fc_w[101,2048]^T via WMMA ----------------
// 1 block, 7 waves; wave nt covers output cols nt*16..nt*16+15 (tail masked at 101)
__global__ void __launch_bounds__(224)
k_fc(const float* __restrict__ feat, const float* __restrict__ fcw,
     float* __restrict__ out, int B) {
  int lane  = threadIdx.x & 31;
  int nt    = threadIdx.x >> 5;            // 0..6
  int b     = lane & 15;                   // A row = graph index
  int ko    = (lane < 16) ? 0 : 8;         // unused; A loads split at kbase too
  int kbase = (lane < 16) ? 0 : 16;        // B-fragment K split
  int n     = nt * 16 + (lane & 15);       // output column

  v8f acc = {0.f, 0.f, 0.f, 0.f, 0.f, 0.f, 0.f, 0.f};
  for (int kb = 0; kb < 64; ++kb) {
    float ta[16];
    if (b < B) {
      const float* ap = feat + (size_t)b * 2048 + kb * 32;
#pragma unroll
      for (int j = 0; j < 8; ++j) { ta[j] = ap[kbase + j]; ta[8 + j] = ap[kbase + 8 + j]; }
    } else {
#pragma unroll
      for (int j = 0; j < 16; ++j) ta[j] = 0.f;
    }
    float tb[16];
    if (n < 101) {
      const float* bp = fcw + (size_t)n * 2048 + kb * 32 + kbase;
#pragma unroll
      for (int j = 0; j < 16; ++j) tb[j] = bp[j];
    } else {
#pragma unroll
      for (int j = 0; j < 16; ++j) tb[j] = 0.f;
    }
    v16h av, bv;
#pragma unroll
    for (int j = 0; j < 16; ++j) { av[j] = (_Float16)ta[j]; bv[j] = (_Float16)tb[j]; }
    acc = __builtin_amdgcn_wmma_f32_16x16x32_f16(false, av, false, bv, (short)0, acc, false, false);
  }
  int mofs = (lane >> 4) * 8;
#pragma unroll
  for (int v = 0; v < 8; ++v) {
    int g = v + mofs;
    if (g < B && n < 101) out[g * 101 + n] = acc[v];
  }
}

// ---------------- host launcher ----------------
extern "C" void kernel_launch(void* const* d_in, const int* in_sizes, int n_in,
                              void* d_out, int out_size, void* d_ws, size_t ws_size,
                              hipStream_t stream) {
  const float* x    = (const float*)d_in[0];
  const float* pos  = (const float*)d_in[1];
  const float* ea   = (const float*)d_in[2];
  const float* W1   = (const float*)d_in[3];
  const float* g1   = (const float*)d_in[4];
  const float* be1  = (const float*)d_in[5];
  const float* W2   = (const float*)d_in[6];
  const float* g2   = (const float*)d_in[7];
  const float* be2  = (const float*)d_in[8];
  const float* fcw  = (const float*)d_in[9];
  const int*   ei   = (const int*)d_in[10];
  const int*   batch = (const int*)d_in[11];

  int N = in_sizes[0];         // x is [N,1]
  int E = in_sizes[2] / 3;     // edge_attr is [E,3]
  int B = out_size / 101;      // out is [B,101]

  char* ws = (char*)d_ws;
  size_t off = 0;
  auto alloc = [&](size_t bytes) -> char* {
    char* p = ws + off;
    off += (bytes + 255) & ~(size_t)255;
    return p;
  };
  float*     acc1   = (float*)alloc((size_t)N * KK * 4);
  float*     deg    = (float*)alloc((size_t)N * 4);
  float*     h1     = (float*)alloc((size_t)N * 32 * 4);
  float*     bn1s   = (float*)alloc(32 * 4);
  float*     bn1q   = (float*)alloc(32 * 4);
  float*     a1     = (float*)alloc(32 * 4);
  float*     b1     = (float*)alloc(32 * 4);
  _Float16*  W2h    = (_Float16*)alloc((size_t)KK * 32 * 128 * 2);
  _Float16*  acc2   = (_Float16*)alloc((size_t)N * KK * 32 * 2);   // packed-f16 accumulator
  float*     h2     = (float*)alloc((size_t)N * 128 * 4);
  float*     bn2s   = (float*)alloc(128 * 4);
  float*     bn2q   = (float*)alloc(128 * 4);
  float*     a2     = (float*)alloc(128 * 4);
  float*     b2     = (float*)alloc(128 * 4);
  float*     pooled = (float*)alloc((size_t)B * 16 * 128 * 4);

  hipMemsetAsync(acc1, 0, (size_t)N * KK * 4, stream);
  hipMemsetAsync(deg,  0, (size_t)N * 4, stream);
  hipMemsetAsync(bn1s, 0, 32 * 4, stream);
  hipMemsetAsync(bn1q, 0, 32 * 4, stream);
  hipMemsetAsync(acc2, 0, (size_t)N * KK * 32 * 2, stream);
  hipMemsetAsync(bn2s, 0, 128 * 4, stream);
  hipMemsetAsync(bn2q, 0, 128 * 4, stream);
  hipMemsetAsync(pooled, 0xFF, (size_t)B * 16 * 128 * 4, stream);  // -NaN bits for atomic max

  k_spline1<<<(E + 255) / 256, 256, 0, stream>>>(ea, x, ei, acc1, deg, E);
  k_node1<<<(N + 7) / 8, 256, 0, stream>>>(acc1, deg, W1, h1, bn1s, bn1q, N);
  k_bn_fin<<<1, 32, 0, stream>>>(bn1s, bn1q, g1, be1, a1, b1, 32, 1.f / (float)N);
  k_bn_apply<<<(N * 32 + 255) / 256, 256, 0, stream>>>(h1, a1, b1, N * 32, 31);
  k_w2pack<<<(KK * 32 * 128 + 255) / 256, 256, 0, stream>>>(W2, W2h);
  {
    long long tthreads = (long long)E * 16;   // 2 edges per wave
    int blocks = (int)((tthreads + 255) / 256);
    k_spline2<<<blocks, 256, 0, stream>>>(ea, h1, ei, acc2, E);
  }
  k_gemm2<<<(N + 63) / 64, 128, 0, stream>>>(acc2, W2h, deg, h2, bn2s, bn2q, N);
  k_bn_fin<<<1, 128, 0, stream>>>(bn2s, bn2q, g2, be2, a2, b2, 128, 1.f / (float)N);
  {
    long long tthreads = (long long)N * 128;
    int blocks = (int)((tthreads + 255) / 256);
    k_pool<<<blocks, 256, 0, stream>>>(h2, a2, b2, pos, batch, pooled, N);
  }
  k_poolfix<<<(B * 2048 + 255) / 256, 256, 0, stream>>>(pooled, B * 2048);
  k_fc<<<1, 224, 0, stream>>>(pooled, fcw, (float*)d_out, B);
}
